// GIFFLARPooling_30236569763927
// MI455X (gfx1250) — matrix operand order repaired and checked by reading, not compiled, not run on hardware
//
#include <hip/hip_runtime.h>

#define D_DIM 1024
#define K2    512            // D_DIM/2 bf16 pairs per row
#define B_SEG 256

typedef __attribute__((ext_vector_type(16))) __bf16 v16bf;
typedef __attribute__((ext_vector_type(8)))  float  v8f;

union Frag { v16bf v; unsigned int u32[8]; uint4 u4[2]; };

// round-to-nearest-even f32 -> bf16, packed pair (lo in [15:0], hi in [31:16])
__device__ __forceinline__ unsigned int pack_bf16(float lo, float hi) {
  unsigned int a = __float_as_uint(lo), b = __float_as_uint(hi);
  a = (a + 0x7FFFu + ((a >> 16) & 1u)) >> 16;
  b = (b + 0x7FFFu + ((b >> 16) & 1u)) & 0xFFFF0000u;
  return a | b;
}

// ---- CDNA5 async global->LDS b128 copy (ASYNCcnt) with sync fallback ----
#if __has_builtin(__builtin_amdgcn_global_load_async_to_lds_b128)
#define ASYNC_LDS 1
#endif

typedef int v4i __attribute__((vector_size(16)));
typedef __attribute__((address_space(1))) v4i gv4i;   // global int4
typedef __attribute__((address_space(3))) v4i lv4i;   // LDS int4

__device__ __forceinline__ void copy_b128_to_lds(const unsigned int* g, unsigned int* l) {
#ifdef ASYNC_LDS
  __builtin_amdgcn_global_load_async_to_lds_b128((gv4i*)g, (lv4i*)l, 0, 0);
#else
  *(uint4*)l = *(const uint4*)g;
#endif
}

__device__ __forceinline__ void wait_async_copies() {
#ifdef ASYNC_LDS
#if __has_builtin(__builtin_amdgcn_s_wait_asynccnt)
  __builtin_amdgcn_s_wait_asynccnt(0);
#else
  asm volatile("s_wait_asynccnt 0" ::: "memory");
#endif
#endif
}

// ---------------- zero output accumulator ----------------
__global__ void zero_out_kernel(float* __restrict__ out, int n) {
  int i = blockIdx.x * blockDim.x + threadIdx.x;
  if (i < n) out[i] = 0.f;
}

// ---- Wn [k][n] fp32 -> wpack [n][k2] bf16-pair uints (WMMA B fragment order) ----
__global__ void pack_w_kernel(const float* __restrict__ Wn, unsigned int* __restrict__ wpack) {
  int idx = blockIdx.x * blockDim.x + threadIdx.x;      // 512*1024 threads
  int k2 = idx >> 10, n = idx & (D_DIM - 1);            // n fast -> coalesced reads
  float lo = Wn[(size_t)(2 * k2) * D_DIM + n];
  float hi = Wn[(size_t)(2 * k2 + 1) * D_DIM + n];
  wpack[(size_t)n * K2 + k2] = pack_bf16(lo, hi);
}

// ---- gate = prelu(x@Wg+bg) fused with x -> xpack [row][k2] bf16 pairs ----
__global__ __launch_bounds__(256) void gate_pack_kernel(
    const float* __restrict__ x, const float* __restrict__ Wg,
    const float* __restrict__ bg, const float* __restrict__ ag,
    float* __restrict__ gate, unsigned int* __restrict__ xpack, int N) {
  const int wave = threadIdx.x >> 5, lane = threadIdx.x & 31;
  const int row = blockIdx.x * 8 + wave;
  if (row >= N) return;
  const float* xr = x + (size_t)row * D_DIM;
  unsigned int* xp = xpack + (size_t)row * K2;
  float p = 0.f;
  #pragma unroll 4
  for (int k2 = lane; k2 < K2; k2 += 32) {
    const float2 v = *(const float2*)(xr + 2 * k2);
    const float2 w = *(const float2*)(Wg + 2 * k2);
    p += v.x * w.x + v.y * w.y;
    xp[k2] = pack_bf16(v.x, v.y);
  }
  for (int off = 16; off; off >>= 1) p += __shfl_xor(p, off, 32);
  if (lane == 0) {
    float g = p + bg[0];
    gate[row] = (g >= 0.f) ? g : ag[0] * g;
  }
}

// ---------------- per-segment max + sum(exp) : one wave per segment ----------------
__global__ void segstats_kernel(const float* __restrict__ gate,
                                const int* __restrict__ batch,
                                float* __restrict__ m, float* __restrict__ s, int N) {
  const int b = blockIdx.x, lane = threadIdx.x;
  int lo = 0, hi = N;
  while (lo < hi) { int mid = (lo + hi) >> 1; if (batch[mid] < b) lo = mid + 1; else hi = mid; }
  int lo2 = lo, hi2 = N;
  while (lo2 < hi2) { int mid = (lo2 + hi2) >> 1; if (batch[mid] < b + 1) lo2 = mid + 1; else hi2 = mid; }
  float mx = -3.0e38f;
  for (int i = lo + lane; i < lo2; i += 32) mx = fmaxf(mx, gate[i]);
  for (int off = 16; off; off >>= 1) mx = fmaxf(mx, __shfl_xor(mx, off, 32));
  float sm = 0.f;
  for (int i = lo + lane; i < lo2; i += 32) sm += __expf(gate[i] - mx);
  for (int off = 16; off; off >>= 1) sm += __shfl_xor(sm, off, 32);
  if (lane == 0) { m[b] = mx; s[b] = sm; }
}

// ---------------- alpha (in place over gate), pre-scaled by 1/3 ----------------
__global__ void alpha_kernel(float* __restrict__ ga, const int* __restrict__ batch,
                             const float* __restrict__ m, const float* __restrict__ s, int N) {
  int i = blockIdx.x * blockDim.x + threadIdx.x;
  if (i >= N) return;
  int b = batch[i];
  ga[i] = __expf(ga[i] - m[b]) / (s[b] + 1e-16f) * (1.f / 3.f);
}

// ---- fused GEMM (bf16 WMMA, pre-packed operands) + prelu + alpha scatter-add ----
// 512 threads = 16 wave32 as 8 row-groups x 2 col-groups; block tile 128x256;
// each wave: 16 rows x 128 cols = 8 accumulators. A frags straight from global
// (2x b128/K-step), B slab double-buffered in LDS via async copy, frags 2x ds b128.
__global__ __launch_bounds__(512) void gemm_pool_kernel(
    const unsigned int* __restrict__ xpack, const int* __restrict__ batch,
    const unsigned int* __restrict__ wpack, const float* __restrict__ bn,
    const float* __restrict__ an, const float* __restrict__ alpha,
    float* __restrict__ out, int N) {
  __shared__ unsigned int ldsB[2][256 * 16];   // [buf][n_local*16 + k2loc], 16KB each
  __shared__ float alpha_sh[128];
  __shared__ int   batch_sh[128];

  const int tid  = threadIdx.x;
  const int lane = tid & 31;
  const int wv   = tid >> 5;
  const int wr   = wv >> 1;          // row group 0..7
  const int wc   = wv & 1;           // col group 0..1
  const int row0 = blockIdx.x * 128;
  const int col0 = blockIdx.y * 256;
  const int m    = lane & 15;
  const int khi  = lane >> 4;

  if (tid < 128) { alpha_sh[tid] = alpha[row0 + tid]; batch_sh[tid] = batch[row0 + tid]; }

  // per-lane A row base (row = row0 + wr*16 + m), already bf16-pair packed
  const unsigned int* aRow = xpack + (size_t)(row0 + wr * 16 + m) * K2;

  // stage one K-slab of B: 256 cols x 16 uints (64B per col); 512 thr x 2 b128
  auto stageB = [&](int kk, int buf) {
    #pragma unroll
    for (int q = 0; q < 2; q++) {
      const int flat = tid * 2 + q;
      const int nl   = flat >> 2;
      const int part = (flat & 3) * 4;
      copy_b128_to_lds(wpack + (size_t)(col0 + nl) * K2 + (kk >> 1) + part,
                       &ldsB[buf][nl * 16 + part]);
    }
  };

  v8f acc[8] = {v8f{}, v8f{}, v8f{}, v8f{}, v8f{}, v8f{}, v8f{}, v8f{}};

  stageB(0, 0);
  wait_async_copies();
  __syncthreads();

  for (int kk = 0; kk < D_DIM; kk += 32) {
    const int buf = (kk >> 5) & 1;
    if (kk + 32 < D_DIM) stageB(kk + 32, buf ^ 1);

    // A fragment: slots 0-3 -> k2 = kk/2 + khi*4 + {0..3}; slots 4-7 -> +8
    Frag a;
    a.u4[0] = *(const uint4*)(aRow + (kk >> 1) + khi * 4);
    a.u4[1] = *(const uint4*)(aRow + (kk >> 1) + 8 + khi * 4);

    #pragma unroll
    for (int t = 0; t < 8; t++) {
      const int nl = wc * 128 + (t << 4) + m;      // col within block tile
      Frag b;                                      // slots 0-3 -> k2loc = khi*4 + v; slots 4-7 -> +8
      b.u4[0] = *(const uint4*)&ldsB[buf][nl * 16 + khi * 4];
      b.u4[1] = *(const uint4*)&ldsB[buf][nl * 16 + 8 + khi * 4];
      acc[t] = __builtin_amdgcn_wmma_f32_16x16x32_bf16(
          false, a.v, false, b.v, (short)0, acc[t], false, false);
    }
    wait_async_copies();
    __syncthreads();
  }

  // epilogue: h = prelu(acc+bn)*alpha; in-tile segment reduce; atomic scatter-add
  const float an0 = an[0];
  const int rbase = wr * 16;
  const bool single_seg = (batch_sh[rbase] == batch_sh[rbase + 15]);
  #pragma unroll
  for (int t = 0; t < 8; t++) {
    const int col = col0 + wc * 128 + (t << 4) + m;
    const float bnv = bn[col];
    if (single_seg) {
      float csum = 0.f;
      #pragma unroll
      for (int r = 0; r < 8; r++) {
        const int rr = rbase + r + (khi << 3);     // C/D: M = r + 8*(lane>>4)
        float h = acc[t][r] + bnv;
        h = (h >= 0.f) ? h : an0 * h;
        csum += h * alpha_sh[rr];
      }
      csum += __shfl_xor(csum, 16, 32);            // lane-halves share the same col
      if (lane < 16)
        atomicAdd(&out[(size_t)batch_sh[rbase] * D_DIM + col], csum);
    } else {
      #pragma unroll
      for (int r = 0; r < 8; r++) {
        const int rr = rbase + r + (khi << 3);
        float h = acc[t][r] + bnv;
        h = (h >= 0.f) ? h : an0 * h;
        atomicAdd(&out[(size_t)batch_sh[rr] * D_DIM + col], h * alpha_sh[rr]);
      }
    }
  }
}

extern "C" void kernel_launch(void* const* d_in, const int* in_sizes, int n_in,
                              void* d_out, int out_size, void* d_ws, size_t ws_size,
                              hipStream_t stream) {
  float* out = (float*)d_out;
  zero_out_kernel<<<(B_SEG * D_DIM + 255) / 256, 256, 0, stream>>>(out, B_SEG * D_DIM);

  // ws layout (buffers reused across the three sequential node types):
  //   xpack : 65536*512 uints = 128 MB
  //   wpack : 512*1024 uints  =   2 MB
  //   gate  : 65536 floats, then m/s (256 each)
  unsigned int* xpack = (unsigned int*)d_ws;
  unsigned int* wpack = xpack + (size_t)65536 * K2;
  float*        gate  = (float*)(wpack + (size_t)K2 * D_DIM);
  float*        m     = gate + 65536;
  float*        s     = m + B_SEG;

  for (int t = 0; t < 3; t++) {
    const float* x  = (const float*)d_in[t * 8 + 0];
    const int*   bt = (const int*)  d_in[t * 8 + 1];
    const float* Wg = (const float*)d_in[t * 8 + 2];
    const float* bg = (const float*)d_in[t * 8 + 3];
    const float* ag = (const float*)d_in[t * 8 + 4];
    const float* Wn = (const float*)d_in[t * 8 + 5];
    const float* bn = (const float*)d_in[t * 8 + 6];
    const float* an = (const float*)d_in[t * 8 + 7];
    const int N = in_sizes[t * 8] / D_DIM;

    pack_w_kernel   <<<(K2 * D_DIM) / 256, 256, 0, stream>>>(Wn, wpack);
    gate_pack_kernel<<<(N + 7) / 8, 256, 0, stream>>>(x, Wg, bg, ag, gate, xpack, N);
    segstats_kernel <<<B_SEG, 32, 0, stream>>>(gate, bt, m, s, N);
    alpha_kernel    <<<(N + 255) / 256, 256, 0, stream>>>(gate, bt, m, s, N);
    gemm_pool_kernel<<<dim3(N / 128, D_DIM / 256), 512, 0, stream>>>(
        xpack, bt, wpack, bn, an, gate, out, N);
  }
}
